// QGCost_23235773072014
// MI455X (gfx1250) — compile-verified, matrix operand order checked by reading, not run on hardware
//
#include <hip/hip_runtime.h>
#include <cstddef>

// ---------------- problem constants ----------------
#define NV 1025            // vertices per side
#define NI 1023            // interior points per side
#define NP 1024            // padded interior (GEMM-friendly), pad row/col = 0

static constexpr float PIF        = 3.14159265358979323846f;
static constexpr float LDOM       = 5120000.0f;
static constexpr float DX         = 5000.0f;             // LDOM / 1024
static constexpr float INV2DX     = 1.0f / (2.0f * DX);
static constexpr float INVDX2     = 1.0f / (DX * DX);
static constexpr float F0C        = 9.375e-05f;
static constexpr float BETA       = 1.754e-11f;
static constexpr float TAU0       = 2e-08f;
static constexpr float DRAG       = 3.60577e-08f;
static constexpr float GC         = 10.0f;
static constexpr float LAM        = (F0C * F0C) / (0.025f * 1000.0f);   // 3.515625e-10
static constexpr float SCALE_PSI  = (GC / F0C) * 0.25f;                 // ssh -> psi (w/ 1/4 factor)
static constexpr float SCALE_BACK = (F0C / GC) / 0.25f;                 // psi -> ssh
static constexpr float DST_NORM   = 0.04419417382415922f;               // sqrt(2/1024)
static constexpr int   N_STEPS    = 120;

typedef __attribute__((ext_vector_type(16))) __bf16        v16bf;
typedef __attribute__((ext_vector_type(8)))  float         v8f;
typedef __attribute__((ext_vector_type(4)))  unsigned int  v4u;
typedef __attribute__((ext_vector_type(8)))  int           v8i_t;
typedef __attribute__((ext_vector_type(4)))  int           v4i_t;

union FragU { v16bf v; uint4 u[2]; };

// ---------------- setup kernels ----------------
__global__ void k_setup_scal(float* scal) {
  if (threadIdx.x == 0) { scal[0] = 0.0f; scal[1] = 0.0f; scal[2] = 0.0f; }
}

// DST-I matrix, bf16 hi/lo split, padded 1024x1024 (row/col 1023 zero).
// arg reduced exactly: sin(pi*(j+1)(k+1)/1024) has integer period 2048 in t=(j+1)(k+1)
__global__ void k_build_S(__bf16* __restrict__ Shi, __bf16* __restrict__ Slo) {
  int idx = blockIdx.x * blockDim.x + threadIdx.x;
  if (idx >= NP * NP) return;
  int j = idx / NP, k = idx % NP;
  float s = 0.0f;
  if (j < NI && k < NI) {
    unsigned t = ((unsigned)(j + 1) * (unsigned)(k + 1)) & 2047u;
    s = DST_NORM * sinf(PIF * (float)t * (1.0f / 1024.0f));
  }
  __bf16 hi = (__bf16)s;
  Shi[idx] = hi;
  Slo[idx] = (__bf16)(s - (float)hi);
}

__global__ void k_build_invH(float* __restrict__ invH) {
  int idx = blockIdx.x * blockDim.x + threadIdx.x;
  if (idx >= NP * NP) return;
  int i = idx / NP, j = idx % NP;
  float v = 0.0f;
  if (i < NI && j < NI) {
    float cx = cosf(PIF * (float)(i + 1) * (1.0f / 1024.0f));
    float cy = cosf(PIF * (float)(j + 1) * (1.0f / 1024.0f));
    float helm = 2.0f * (cx - 1.0f) * INVDX2 + 2.0f * (cy - 1.0f) * INVDX2 - LAM;
    v = 1.0f / helm;
  }
  invH[idx] = v;
}

__global__ void k_build_wind(float* __restrict__ wind) {
  int j = blockIdx.x * blockDim.x + threadIdx.x;
  if (j >= NI) return;
  float y = -0.5f * LDOM + (float)(j + 1) * DX;
  float w = 2.0f * PIF / LDOM;
  wind[j] = -TAU0 * w * sinf(w * y);
}

// psi0 = SCALE_PSI * ssh0, ssh0 = state[0,0].T   (psi[i*NV+j] indexed as [x][y])
__global__ void k_psi0(float* __restrict__ psi, const float* __restrict__ state) {
  int idx = blockIdx.x * blockDim.x + threadIdx.x;
  if (idx >= NV * NV) return;
  int i = idx / NV, j = idx % NV;
  psi[idx] = SCALE_PSI * state[j * NV + i];
}

__global__ void k_q0(float* __restrict__ q, const float* __restrict__ psi) {
  int idx = blockIdx.x * blockDim.x + threadIdx.x;
  if (idx >= NP * NP) return;
  int i = idx / NP, j = idx % NP;
  float val = 0.0f;
  if (i < NI && j < NI) {
    int c = (i + 1) * NV + (j + 1);
    float pc  = psi[c];
    float lap = (psi[c + NV] - 2.0f * pc + psi[c - NV]) * INVDX2
              + (psi[c + 1]  - 2.0f * pc + psi[c - 1])  * INVDX2;
    val = lap - LAM * pc;
  }
  q[idx] = val;
}

// |u|,|v| maxima via atomicMax on non-negative float bit patterns
__global__ void k_vmax(const float* __restrict__ psi, float* scal) {
  int idx = blockIdx.x * blockDim.x + threadIdx.x;
  const int NU = NV * (NV - 1);
  float av; int slot;
  if (idx < NU) {
    int x = idx / (NV - 1), yj = idx % (NV - 1);
    av = fabsf((psi[x * NV + yj + 1] - psi[x * NV + yj]) * (1.0f / DX));
    slot = 0;
  } else if (idx < 2 * NU) {
    int t = idx - NU;
    int xi = t / NV, y = t % NV;
    av = fabsf((psi[(xi + 1) * NV + y] - psi[xi * NV + y]) * (1.0f / DX));
    slot = 1;
  } else return;
  atomicMax((int*)scal + slot, __float_as_int(av));
}

__global__ void k_dt(float* scal) {
  if (threadIdx.x == 0 && blockIdx.x == 0) {
    float um = __int_as_float(((int*)scal)[0]); if (um == 0.0f) um = 4.0f;
    float vm = __int_as_float(((int*)scal)[1]); if (vm == 0.0f) vm = 4.0f;
    scal[2] = fminf(DX / um, DX / vm);          // CFL = 1
  }
}

// qout = c0*qbase + c1*qB + c2*dt*rhs(qB, psi)  on padded NP grid (pads -> 0)
__global__ void k_rhs(float* __restrict__ qout, const float* __restrict__ qbase,
                      const float* __restrict__ qB, const float* __restrict__ psi,
                      const float* __restrict__ wind, const float* __restrict__ scal,
                      float c0, float c1, float c2) {
  int idx = blockIdx.x * blockDim.x + threadIdx.x;
  if (idx >= NP * NP) return;
  int i = idx / NP, j = idx % NP;
  if (i >= NI || j >= NI) { qout[idx] = 0.0f; return; }
  float dt = scal[2];
  int c = (i + 1) * NV + (j + 1);
  float pc = psi[c];
  float pE = psi[c + NV], pW = psi[c - NV];
  float pN = psi[c + 1],  pS = psi[c - 1];
  float u = -(pN - pS) * INV2DX;
  float v =  (pE - pW) * INV2DX;
  float qxp = qB[idx + NP];                          // row 1023 is zero pad
  float qxm = (i > 0) ? qB[idx - NP] : 0.0f;
  float qyp = qB[idx + 1];                           // col 1023 is zero pad
  float qym = (j > 0) ? qB[idx - 1] : 0.0f;
  float qx  = (qxp - qxm) * INV2DX;
  float qy  = (qyp - qym) * INV2DX;
  float lap = (pE - 2.0f * pc + pW) * INVDX2 + (pN - 2.0f * pc + pS) * INVDX2;
  float rhs = -(u * qx + v * qy) - BETA * v + wind[j] - DRAG * lap;
  qout[idx] = c0 * qbase[idx] + c1 * qB[idx] + c2 * dt * rhs;
}

// fp32 -> bf16 hi/lo split; optional elementwise *invH (Helmholtz divide, H symmetric)
// optional transpose so every GEMM is plain C = S * B
__global__ void k_conv(const float* __restrict__ src, __bf16* __restrict__ bhi,
                       __bf16* __restrict__ blo, const float* __restrict__ invH, int trans) {
  int idx = blockIdx.x * blockDim.x + threadIdx.x;
  if (idx >= NP * NP) return;
  float v = src[idx];
  if (invH) v *= invH[idx];
  int o = trans ? ((idx % NP) * NP + (idx / NP)) : idx;
  __bf16 h = (__bf16)v;
  bhi[o] = h;
  blo[o] = (__bf16)(v - (float)h);
}

// ---------------- TDM: DMA one 128x32 bf16 slab (row-major, stride 1024) into LDS ---
// D# pad fields synthesize the 40-element (80B) LDS pitch: 64B row (pad_interval=3:
// 16 DWORDs) + 16B pad (pad_amount=3: 4 DWORDs). Tracked by TENSORcnt.
__device__ __forceinline__ void tdm_load_tile_A(const __bf16* gptr, unsigned lds_byte_off) {
  unsigned long long ga = (unsigned long long)(const void*)gptr;
  v4u g0;
  g0[0] = 1u;                                              // count=1, user descriptor
  g0[1] = lds_byte_off;                                    // LDS dest (bytes)
  g0[2] = (unsigned)ga;                                    // global_addr[31:0]
  g0[3] = (unsigned)((ga >> 32) & 0x01FFFFFFull) | (2u << 30);  // addr[56:32] | type=2
  v8i_t g1;
  g1[0] = (int)((1u << 16)      // data_size = 2 bytes (bf16)
              | (1u << 20)      // pad_enable
              | (3u << 22)      // pad_interval: 16 DWORDs (64B) between pads
              | (3u << 25));    // pad_amount: 4 DWORDs (16B)
  g1[1] = (int)(1024u << 16);   // tensor_dim0 = 1024 (bits 79:48)
  g1[2] = (int)(1024u << 16);   // tensor_dim1 = 1024 (bits 111:80)
  g1[3] = (int)(32u << 16);     // tile_dim0 = 32 (K slab width)
  g1[4] = (int)128;             // tile_dim1 = 128 rows, tile_dim2 = 0
  g1[5] = (int)1024;            // tensor_dim0_stride = 1024 elements
  g1[6] = 0;
  g1[7] = 0;
  v4i_t gz = {0, 0, 0, 0};
#if __clang_major__ >= 23
  v8i_t gz8 = {0, 0, 0, 0, 0, 0, 0, 0};
  __builtin_amdgcn_tensor_load_to_lds(g0, g1, gz, gz, gz8, 0);
#else
  __builtin_amdgcn_tensor_load_to_lds(g0, g1, gz, gz, 0);
#endif
}

// ---------------- WMMA GEMM: C(f32) = (Ahi+Alo)(Bhi+Blo), 1024^3, S on the left ---
// block tile 128x128, 8 waves of 32x64, bf16 split => 3 x v_wmma_f32_16x16x32_bf16
__launch_bounds__(256)
__global__ void k_gemm(const __bf16* __restrict__ Ahi, const __bf16* __restrict__ Alo,
                       const __bf16* __restrict__ Bhi, const __bf16* __restrict__ Blo,
                       float* __restrict__ C) {
  __shared__ unsigned short Ash_h[128 * 40], Ash_l[128 * 40];
  __shared__ unsigned short Bsh_h[128 * 40], Bsh_l[128 * 40];   // stored transposed [n][k]

  const int tid  = threadIdx.x;
  const int lane = tid & 31, wave = tid >> 5;
  const int waveM = wave & 3, waveN = wave >> 2;
  const int mbase = blockIdx.x * 128;
  const int nbase = blockIdx.y * 128;

  // LDS aperture is 4GB-aligned: low 32 bits of a generic pointer to __shared__
  // are the raw LDS byte offset.
  const unsigned ash_h_off = (unsigned)(size_t)(const void*)Ash_h;
  const unsigned ash_l_off = (unsigned)(size_t)(const void*)Ash_l;

  v8f acc[2][4];
  v8f zero = {};
  #pragma unroll
  for (int a = 0; a < 2; a++)
    #pragma unroll
    for (int b = 0; b < 4; b++) acc[a][b] = zero;

  for (int kc = 0; kc < NP; kc += 32) {
    // prefetch next K-slab of B while this one is consumed (global_prefetch_b8)
    if (kc + 32 < NP) {
      const __bf16* pb = (tid & 1) ? Blo : Bhi;
      __builtin_prefetch(pb + (size_t)(kc + 32 + (tid >> 3)) * NP + nbase + (tid & 7) * 16, 0, 1);
    }
    __syncthreads();                       // everyone done reading previous slabs

    // A tile (128 x 32 bf16, hi & lo planes) via Tensor Data Mover, wave 0 issues
    if (wave == 0) {
      const size_t goff = (size_t)mbase * NP + kc;
      tdm_load_tile_A(Ahi + goff, ash_h_off);
      tdm_load_tile_A(Alo + goff, ash_l_off);
    }

    // B tile (32 x 128) staged transposed into [n][k] so fragments read rows
    #pragma unroll
    for (int x = tid; x < 512; x += 256) {
      int k = x >> 4, n0 = (x & 15) * 8;
      size_t g = (size_t)(kc + k) * NP + nbase + n0;
      uint4 bh = *(const uint4*)(Bhi + g);
      uint4 bl = *(const uint4*)(Blo + g);
      const unsigned short* ph = (const unsigned short*)&bh;
      const unsigned short* pl = (const unsigned short*)&bl;
      #pragma unroll
      for (int e = 0; e < 8; e++) {
        Bsh_h[(n0 + e) * 40 + k] = ph[e];
        Bsh_l[(n0 + e) * 40 + k] = pl[e];
      }
    }

    if (wave == 0) __builtin_amdgcn_s_wait_tensorcnt(0);   // A slab landed in LDS
    __syncthreads();                                       // publish A+B to all waves

    // build fragments: 16-bit A-matrix layout — lane<16: K {0..7,16..23}; lane>=16: K {8..15,24..31}
    const int r = lane & 15;
    const int grp = (lane >> 4) * 8;
    FragU fa_h[2], fa_l[2], fb_h[4], fb_l[4];
    #pragma unroll
    for (int mt = 0; mt < 2; mt++) {
      int row = waveM * 32 + mt * 16 + r;
      fa_h[mt].u[0] = *(const uint4*)&Ash_h[row * 40 + grp];
      fa_h[mt].u[1] = *(const uint4*)&Ash_h[row * 40 + grp + 16];
      fa_l[mt].u[0] = *(const uint4*)&Ash_l[row * 40 + grp];
      fa_l[mt].u[1] = *(const uint4*)&Ash_l[row * 40 + grp + 16];
    }
    #pragma unroll
    for (int nt = 0; nt < 4; nt++) {
      int col = waveN * 64 + nt * 16 + r;
      fb_h[nt].u[0] = *(const uint4*)&Bsh_h[col * 40 + grp];
      fb_h[nt].u[1] = *(const uint4*)&Bsh_h[col * 40 + grp + 16];
      fb_l[nt].u[0] = *(const uint4*)&Bsh_l[col * 40 + grp];
      fb_l[nt].u[1] = *(const uint4*)&Bsh_l[col * 40 + grp + 16];
    }

    // split product: hi*hi + hi*lo + lo*hi (f32 accumulate)
    #pragma unroll
    for (int mt = 0; mt < 2; mt++)
      #pragma unroll
      for (int nt = 0; nt < 4; nt++) {
        acc[mt][nt] = __builtin_amdgcn_wmma_f32_16x16x32_bf16(
            false, fa_h[mt].v, false, fb_h[nt].v, (short)0, acc[mt][nt], false, false);
        acc[mt][nt] = __builtin_amdgcn_wmma_f32_16x16x32_bf16(
            false, fa_h[mt].v, false, fb_l[nt].v, (short)0, acc[mt][nt], false, false);
        acc[mt][nt] = __builtin_amdgcn_wmma_f32_16x16x32_bf16(
            false, fa_l[mt].v, false, fb_h[nt].v, (short)0, acc[mt][nt], false, false);
      }
  }

  // epilogue: C layout — VGPR r: lanes 0..15 -> M=r, lanes 16..31 -> M=r+8
  const int rr = lane & 15, half = lane >> 4;
  #pragma unroll
  for (int mt = 0; mt < 2; mt++) {
    int row0 = mbase + waveM * 32 + mt * 16 + half * 8;
    #pragma unroll
    for (int nt = 0; nt < 4; nt++) {
      int col = nbase + waveN * 64 + nt * 16 + rr;
      #pragma unroll
      for (int e = 0; e < 8; e++)
        C[(size_t)(row0 + e) * NP + col] = acc[mt][nt][e];
    }
  }
}

__global__ void k_embed(float* __restrict__ psi, const float* __restrict__ src) {
  int idx = blockIdx.x * blockDim.x + threadIdx.x;
  if (idx >= NI * NI) return;
  int i = idx / NI, j = idx % NI;
  psi[(i + 1) * NV + (j + 1)] = src[i * NP + j];
}

// MSE: forecasts[0]==state[0,0] (zero term); ssh1.T[r][c] = SCALE_BACK * psi[c*NV+r]
__global__ void k_final(const float* __restrict__ state, const float* __restrict__ psi,
                        float* __restrict__ partials) {
  int idx = blockIdx.x * blockDim.x + threadIdx.x;
  float d2 = 0.0f;
  if (idx < NV * NV) {
    int rrow = idx / NV, ccol = idx % NV;
    float ref  = state[NV * NV + idx];
    float pred = SCALE_BACK * psi[ccol * NV + rrow];
    float d = ref - pred;
    d2 = d * d;
  }
  __shared__ float sm[256];
  sm[threadIdx.x] = d2;
  __syncthreads();
  for (int o = 128; o > 0; o >>= 1) {
    if (threadIdx.x < o) sm[threadIdx.x] += sm[threadIdx.x + o];
    __syncthreads();
  }
  if (threadIdx.x == 0) partials[blockIdx.x] = sm[0];
}

__global__ void k_final2(const float* __restrict__ partials, int n, float* __restrict__ out) {
  float s = 0.0f;
  for (int i = threadIdx.x; i < n; i += 256) s += partials[i];
  __shared__ float sm[256];
  sm[threadIdx.x] = s;
  __syncthreads();
  for (int o = 128; o > 0; o >>= 1) {
    if (threadIdx.x < o) sm[threadIdx.x] += sm[threadIdx.x + o];
    __syncthreads();
  }
  if (threadIdx.x == 0) out[0] = sm[0] / (2.0f * (float)(NV * NV));
}

// ---------------- host orchestration ----------------
extern "C" void kernel_launch(void* const* d_in, const int* in_sizes, int n_in,
                              void* d_out, int out_size, void* d_ws, size_t ws_size,
                              hipStream_t stream) {
  (void)in_sizes; (void)n_in; (void)out_size;
  const float* state = (const float*)d_in[0];     // (1,2,1025,1025) f32
  char* w = (char*)d_ws;
  size_t off = 0;
  auto alloc = [&](size_t b) { size_t o = off; off += (b + 255) & ~(size_t)255; return o; };

  float*  psi  = (float*)(w + alloc(sizeof(float) * NV * NV));
  float*  q    = (float*)(w + alloc(sizeof(float) * NP * NP));
  float*  qsA  = (float*)(w + alloc(sizeof(float) * NP * NP));
  float*  qsB  = (float*)(w + alloc(sizeof(float) * NP * NP));
  float*  ta   = (float*)(w + alloc(sizeof(float) * NP * NP));
  float*  tb   = (float*)(w + alloc(sizeof(float) * NP * NP));
  __bf16* Shi  = (__bf16*)(w + alloc(2 * (size_t)NP * NP));
  __bf16* Slo  = (__bf16*)(w + alloc(2 * (size_t)NP * NP));
  __bf16* Bhi  = (__bf16*)(w + alloc(2 * (size_t)NP * NP));
  __bf16* Blo  = (__bf16*)(w + alloc(2 * (size_t)NP * NP));
  float*  invH = (float*)(w + alloc(sizeof(float) * NP * NP));
  float*  wind = (float*)(w + alloc(sizeof(float) * NI));
  float*  scal = (float*)(w + alloc(sizeof(float) * 4));
  float*  part = (float*)(w + alloc(sizeof(float) * 8192));
  if (off > ws_size) return;

  const int TPB = 256;
  const int G_NP2 = (NP * NP + TPB - 1) / TPB;
  const int G_NV2 = (NV * NV + TPB - 1) / TPB;
  const int G_NI2 = (NI * NI + TPB - 1) / TPB;
  const int G_VMX = (2 * NV * (NV - 1) + TPB - 1) / TPB;
  dim3 gemmGrid(NP / 128, NP / 128);

  // setup (recomputed every call: no cached state)
  k_setup_scal<<<1, 32, 0, stream>>>(scal);
  k_build_S<<<G_NP2, TPB, 0, stream>>>(Shi, Slo);
  k_build_invH<<<G_NP2, TPB, 0, stream>>>(invH);
  k_build_wind<<<(NI + TPB - 1) / TPB, TPB, 0, stream>>>(wind);
  k_psi0<<<G_NV2, TPB, 0, stream>>>(psi, state);
  k_q0<<<G_NP2, TPB, 0, stream>>>(q, psi);
  k_vmax<<<G_VMX, TPB, 0, stream>>>(psi, scal);
  k_dt<<<1, 32, 0, stream>>>(scal);

  // solve(q) = S*((S q S) / H)*S via 4 GEMMs (S symmetric; transpose folded into conv)
  auto solve_embed = [&](const float* src) {
    k_conv<<<G_NP2, TPB, 0, stream>>>(src, Bhi, Blo, (const float*)nullptr, 0);
    k_gemm<<<gemmGrid, TPB, 0, stream>>>(Shi, Slo, Bhi, Blo, ta);
    k_conv<<<G_NP2, TPB, 0, stream>>>(ta, Bhi, Blo, (const float*)nullptr, 1);
    k_gemm<<<gemmGrid, TPB, 0, stream>>>(Shi, Slo, Bhi, Blo, tb);   // tb = (S src S)^T
    k_conv<<<G_NP2, TPB, 0, stream>>>(tb, Bhi, Blo, invH, 0);       // divide by H (symmetric)
    k_gemm<<<gemmGrid, TPB, 0, stream>>>(Shi, Slo, Bhi, Blo, ta);
    k_conv<<<G_NP2, TPB, 0, stream>>>(ta, Bhi, Blo, (const float*)nullptr, 1);
    k_gemm<<<gemmGrid, TPB, 0, stream>>>(Shi, Slo, Bhi, Blo, tb);   // tb = solve(src)
    k_embed<<<G_NI2, TPB, 0, stream>>>(psi, tb);
  };

  for (int s = 0; s < N_STEPS; s++) {
    // q1 = q + dt*rhs(q, p)
    k_rhs<<<G_NP2, TPB, 0, stream>>>(qsA, q, q, psi, wind, scal, 1.0f, 0.0f, 1.0f);
    solve_embed(qsA);
    // q2 = 0.75 q + 0.25 q1 + 0.25 dt rhs(q1, p1)
    k_rhs<<<G_NP2, TPB, 0, stream>>>(qsB, q, qsA, psi, wind, scal, 0.75f, 0.25f, 0.25f);
    solve_embed(qsB);
    // qn = q/3 + (2/3) q2 + (2/3) dt rhs(q2, p2)
    k_rhs<<<G_NP2, TPB, 0, stream>>>(q, q, qsB, psi, wind, scal,
                                     1.0f / 3.0f, 2.0f / 3.0f, 2.0f / 3.0f);
    solve_embed(q);
  }

  k_final<<<G_NV2, TPB, 0, stream>>>(state, psi, part);
  k_final2<<<1, 256, 0, stream>>>(part, G_NV2, (float*)d_out);
}
